// Head_57105885167911
// MI455X (gfx1250) — compile-verified, matrix operand order checked by reading, not run on hardware
//
#include <hip/hip_runtime.h>

typedef __attribute__((ext_vector_type(16))) _Float16 v16h;
typedef __attribute__((ext_vector_type(8)))  _Float16 v8h;
typedef __attribute__((ext_vector_type(8)))  float    v8f;

#define BB   256
#define TT   256
#define CEMB 384
#define HD   64
#define SCL  0.05103103630798287f   // 384^-0.5

static __device__ __forceinline__ float hmax16(float v) {
  v = fmaxf(v, __shfl_xor(v, 1, 32));
  v = fmaxf(v, __shfl_xor(v, 2, 32));
  v = fmaxf(v, __shfl_xor(v, 4, 32));
  v = fmaxf(v, __shfl_xor(v, 8, 32));
  return v;
}
static __device__ __forceinline__ float hsum16(float v) {
  v += __shfl_xor(v, 1, 32);
  v += __shfl_xor(v, 2, 32);
  v += __shfl_xor(v, 4, 32);
  v += __shfl_xor(v, 8, 32);
  return v;
}

// Fragment loader: elements 0..7 at p[0..7], elements 8..15 at p[16..23]
// (the 16-bit WMMA A/B per-lane layout). Both halves are 16B-aligned.
static __device__ __forceinline__ v16h load_frag(const _Float16* p) {
  v8h lo = *(const v8h*)p;
  v8h hi = *(const v8h*)(p + 16);
  return __builtin_shufflevector(lo, hi, 0, 1, 2, 3, 4, 5, 6, 7,
                                 8, 9, 10, 11, 12, 13, 14, 15);
}

#define WMMA_F16(A, B, C) \
  __builtin_amdgcn_wmma_f32_16x16x32_f16(false, (A), false, (B), (short)0, (C), false, false)

// ---------------------------------------------------------------------------
// Kernel 0: transpose + convert weights once: Wt[col][k] = (f16) W[k][col].
// Makes B-fragment elements contiguous in memory (two b128 loads per frag).
// ---------------------------------------------------------------------------
__global__ __launch_bounds__(256) void wprep_kernel(
    const float* __restrict__ Wq, const float* __restrict__ Wk,
    const float* __restrict__ Wv,
    _Float16* __restrict__ wtq, _Float16* __restrict__ wtk,
    _Float16* __restrict__ wtv)
{
  const int idx = blockIdx.x * 256 + threadIdx.x;   // 0 .. 64*384-1
  const int col = idx / CEMB;
  const int k   = idx - col * CEMB;
  const int src = k * HD + col;
  wtq[idx] = (_Float16)Wq[src];
  wtk[idx] = (_Float16)Wk[src];
  wtv[idx] = (_Float16)Wv[src];
}

// ---------------------------------------------------------------------------
// Kernel 1: fused QKV projection.  x[65536,384] x W[384,64] -> q,k (f16 row-
// major [B*T,64]) and v transposed (f16 [B,64,T]).
// Grid: 4096 blocks (16 token rows each) x 128 threads; wave w owns 16 cols.
// ---------------------------------------------------------------------------
__global__ __launch_bounds__(128) void qkv_proj_kernel(
    const float* __restrict__ x,
    const _Float16* __restrict__ wtq, const _Float16* __restrict__ wtk,
    const _Float16* __restrict__ wtv,
    _Float16* __restrict__ qf, _Float16* __restrict__ kf,
    _Float16* __restrict__ vt)
{
  const int lane = threadIdx.x & 31;
  const int wave = threadIdx.x >> 5;     // column tile 0..3
  const int rt   = blockIdx.x;           // 16-row tile of B*T
  const int colh = lane & 15;
  const int hi   = lane >> 4;
  const int kb   = hi * 8;
  const int col  = wave * 16 + colh;

  const float* xrow = x + ((size_t)rt * 16 + colh) * CEMB;
  const _Float16* wq = wtq + (size_t)col * CEMB;
  const _Float16* wk = wtk + (size_t)col * CEMB;
  const _Float16* wv = wtv + (size_t)col * CEMB;

  v8f cq = {}; v8f ck = {}; v8f cv = {};

  for (int kk = 0; kk < CEMB; kk += 32) {
    // A fragment: 16x32 chunk of x (f32 -> f16, four b128 loads).
    v16h a;
    const float4* p0 = (const float4*)(xrow + kk + kb);
    const float4* p1 = (const float4*)(xrow + kk + kb + 16);
    float4 x0 = p0[0], x1 = p0[1];
    float4 y0 = p1[0], y1 = p1[1];
    a[0]=(_Float16)x0.x; a[1]=(_Float16)x0.y; a[2]=(_Float16)x0.z; a[3]=(_Float16)x0.w;
    a[4]=(_Float16)x1.x; a[5]=(_Float16)x1.y; a[6]=(_Float16)x1.z; a[7]=(_Float16)x1.w;
    a[8]=(_Float16)y0.x; a[9]=(_Float16)y0.y; a[10]=(_Float16)y0.z; a[11]=(_Float16)y0.w;
    a[12]=(_Float16)y1.x; a[13]=(_Float16)y1.y; a[14]=(_Float16)y1.z; a[15]=(_Float16)y1.w;

    // B fragments: contiguous f16 (two b128 loads each).
    const v16h bq = load_frag(wq + kk + kb);
    const v16h bk = load_frag(wk + kk + kb);
    const v16h bv = load_frag(wv + kk + kb);

    cq = WMMA_F16(a, bq, cq);
    ck = WMMA_F16(a, bk, ck);
    cv = WMMA_F16(a, bv, cv);
  }

#pragma unroll
  for (int j = 0; j < 8; ++j) {
    const int r = rt * 16 + j + hi * 8;            // global token row
    qf[(size_t)r * HD + col] = (_Float16)cq[j];
    kf[(size_t)r * HD + col] = (_Float16)ck[j];
    const int b = r >> 8;                          // r / T
    const int t = r & 255;                         // r % T
    vt[((size_t)b * HD + col) * TT + t] = (_Float16)cv[j];
  }
}

// ---------------------------------------------------------------------------
// Kernel 2: causal flash attention. One wave per (batch, 16-query tile).
// Online softmax over 32-key supertiles; P routed through LDS (C-layout ->
// A-fragment transpose); PV uses transposed-V for contiguous B-fragments.
// Grid: B*(T/64)=1024 blocks x 128 threads (4 independent waves).
// Query tiles are interleaved across waves (qt = wave*4 + g) so every block
// carries a near-constant amount of causal work (16 or 20 supertiles).
// ---------------------------------------------------------------------------
__global__ __launch_bounds__(128) void flash_attn_kernel(
    const _Float16* __restrict__ qf, const _Float16* __restrict__ kf,
    const _Float16* __restrict__ vt, float* __restrict__ out)
{
  __shared__ __align__(16) _Float16 plds[4][16 * 32];

  const int lane   = threadIdx.x & 31;
  const int wave   = threadIdx.x >> 5;
  const int b      = blockIdx.x >> 2;
  const int qt     = wave * 4 + (blockIdx.x & 3); // interleaved query tile
  const int qbase  = qt * 16;
  const int colh   = lane & 15;
  const int hi     = lane >> 4;
  const int kb     = hi * 8;
  const int row_off = hi * 8;

  // Q A-fragments (head dim 0..31 and 32..63), b128 loads.
  const _Float16* qrow = qf + ((size_t)b * TT + qbase + colh) * HD;
  const v16h aq0 = load_frag(qrow + kb);
  const v16h aq1 = load_frag(qrow + 32 + kb);

  float m[8], l[8];
  v8f o[4] = {{}, {}, {}, {}};
#pragma unroll
  for (int j = 0; j < 8; ++j) { m[j] = -1e30f; l[j] = 0.0f; }

  _Float16* P = &plds[wave][0];
  const int nst = qt / 2 + 1;                      // causal supertile count

  for (int st = 0; st < nst; ++st) {
    const int kt0 = 2 * st, kt1 = 2 * st + 1;

    // S tile A: queries x keys [16kt0, 16kt0+16)
    v8f cA = {};
    {
      const _Float16* krow = kf + ((size_t)b * TT + kt0 * 16 + colh) * HD;
      cA = WMMA_F16(aq0, load_frag(krow + kb), cA);
      cA = WMMA_F16(aq1, load_frag(krow + 32 + kb), cA);
    }
    const bool haveB = (kt1 <= qt);                // second subtile live?
    v8f cB = {};
    if (haveB) {
      const _Float16* krow = kf + ((size_t)b * TT + kt1 * 16 + colh) * HD;
      cB = WMMA_F16(aq0, load_frag(krow + kb), cB);
      cB = WMMA_F16(aq1, load_frag(krow + 32 + kb), cB);
    }

    // Scale + causal mask + online softmax update; stage P into LDS.
#pragma unroll
    for (int j = 0; j < 8; ++j) {
      const int qr = qbase + j + row_off;
      float sa = cA[j] * SCL;
      if (kt0 * 16 + colh > qr) sa = -1e30f;
      float sb = haveB ? (cB[j] * SCL) : -1e30f;
      if (kt1 * 16 + colh > qr) sb = -1e30f;

      const float rm  = hmax16(fmaxf(sa, sb));
      const float mn  = fmaxf(m[j], rm);
      const float fac = __expf(m[j] - mn);
      const float pa  = __expf(sa - mn);
      const float pb  = __expf(sb - mn);
      l[j] = l[j] * fac + hsum16(pa + pb);
      m[j] = mn;
      o[0][j] *= fac; o[1][j] *= fac; o[2][j] *= fac; o[3][j] *= fac;

      const int r = j + row_off;
      P[r * 32 + colh]      = (_Float16)pa;
      P[r * 32 + 16 + colh] = (_Float16)pb;
    }

    // Re-load P as a 16x32 A-fragment (same-wave DS ops are in order; the
    // compiler inserts s_wait_dscnt). Two ds_load_b128 per lane.
    const v16h ap = load_frag(P + colh * 32 + kb);

    // O += P @ V  (V transposed: contiguous along key dim, b128 loads).
    const _Float16* vbase = vt + (size_t)b * HD * TT + 32 * st;
#pragma unroll
    for (int t = 0; t < 4; ++t) {
      const _Float16* vcol = vbase + (size_t)(16 * t + colh) * TT;
      o[t] = WMMA_F16(ap, load_frag(vcol + kb), o[t]);
    }
  }

  // Normalize and write fp32 output [B, T, 64].
  float inv[8];
#pragma unroll
  for (int j = 0; j < 8; ++j) inv[j] = 1.0f / l[j];
#pragma unroll
  for (int t = 0; t < 4; ++t) {
#pragma unroll
    for (int j = 0; j < 8; ++j) {
      const int qr = qbase + j + row_off;
      out[((size_t)b * TT + qr) * HD + 16 * t + colh] = o[t][j] * inv[j];
    }
  }
}

extern "C" void kernel_launch(void* const* d_in, const int* in_sizes, int n_in,
                              void* d_out, int out_size, void* d_ws, size_t ws_size,
                              hipStream_t stream) {
  const float* x  = (const float*)d_in[0];
  const float* Wq = (const float*)d_in[1];
  const float* Wk = (const float*)d_in[2];
  const float* Wv = (const float*)d_in[3];
  float* out = (float*)d_out;

  // Workspace: q, k (f16 [B*T,64]), v-transposed (f16 [B,64,T]),
  // three transposed f16 weights ([64,384]).  ~25.3 MB total.
  _Float16* qf  = (_Float16*)d_ws;
  _Float16* kf  = qf + (size_t)BB * TT * HD;
  _Float16* vt  = kf + (size_t)BB * TT * HD;
  _Float16* wtq = vt + (size_t)BB * TT * HD;
  _Float16* wtk = wtq + (size_t)HD * CEMB;
  _Float16* wtv = wtk + (size_t)HD * CEMB;

  wprep_kernel<<<(HD * CEMB) / 256, 256, 0, stream>>>(Wq, Wk, Wv, wtq, wtk, wtv);
  qkv_proj_kernel<<<(BB * TT) / 16, 128, 0, stream>>>(x, wtq, wtk, wtv, qf, kf, vt);
  flash_attn_kernel<<<BB * (TT / 64), 128, 0, stream>>>(qf, kf, vt, out);
}
